// LSTMNet_47399259079068
// MI455X (gfx1250) — compile-verified, matrix operand order checked by reading, not run on hardware
//
#include <hip/hip_runtime.h>

// 2-layer LSTM (H=64) + Linear(64,3)+LeakyReLU head, fully fused.
// B=512, T=2048. One workgroup per 16-row batch tile, 8 waves (256 thr).
// Recurrent + input GEMMs via v_wmma_f32_16x16x32_bf16, f32 accumulate.
// Weights live in VGPRs as preloaded B-fragments; h/c state lives in LDS.

constexpr int H    = 64;
constexpr int DIN  = 12;
constexpr int NO   = 3;
constexpr int BSZ  = 512;
constexpr int TLEN = 2048;
constexpr int BT   = 16;     // batch tile (WMMA M)
constexpr int G4H  = 4 * H;  // 256 gate columns

constexpr float LOG2E = 1.4426950408889634f;

typedef __attribute__((ext_vector_type(16))) __bf16 v16bf;
typedef __attribute__((ext_vector_type(8)))  __bf16 v8bf;
typedef __attribute__((ext_vector_type(8)))  float  v8f;

__device__ __forceinline__ v8f wmma_bf16(v16bf a, v16bf b, v8f c) {
  // D = A(16x32 bf16) * B(32x16 bf16) + C(16x16 f32)
  return __builtin_amdgcn_wmma_f32_16x16x32_bf16(false, a, false, b, (short)0, c,
                                                 false, false);
}

// Raw hardware transcendentals: single v_rcp_f32 / v_exp_f32.
__device__ __forceinline__ float fast_rcp(float x)  { return __builtin_amdgcn_rcpf(x); }
__device__ __forceinline__ float fast_exp2(float x) { return __builtin_amdgcn_exp2f(x); }

// tanh(x) = 2*rcp(1+exp2(-2*log2e*x)) - 1 : 1 mul + 1 add + 1 fma + 2 TRANS.
__device__ __forceinline__ float tanh_fast(float x) {
  return fmaf(2.0f, fast_rcp(1.0f + fast_exp2(-2.0f * LOG2E * x)), -1.0f);
}

// B-fragment (K=32 x N=16 chunk of B[k][n] = W[n][k]) in the wave32 16-bit layout:
// lane<16: n=n0+lane, halfs[0..7]=B[k0+0..7][n],  halfs[8..15]=B[k0+16..23][n]
// lane>=16: same n,   halfs[0..7]=B[k0+8..15][n], halfs[8..15]=B[k0+24..31][n]
__device__ __forceinline__ v16bf load_bfrag(const float* __restrict__ W, int K,
                                            int n0, int k0, int lane) {
  const int n    = n0 + (lane & 15);
  const int koff = (lane & 16) ? 8 : 0;
  const float* row = W + n * K;            // W row n is contiguous in k
  v16bf b;
#pragma unroll
  for (int j = 0; j < 8; ++j) {
    int k = k0 + koff + j;
    b[j] = (k < K) ? (__bf16)row[k] : (__bf16)0.0f;
  }
#pragma unroll
  for (int j = 0; j < 8; ++j) {
    int k = k0 + 16 + koff + j;
    b[8 + j] = (k < K) ? (__bf16)row[k] : (__bf16)0.0f;
  }
  return b;
}

// A-fragment (16xK=32 chunk) from the bf16 hidden state in LDS (row = lane&15).
__device__ __forceinline__ v16bf load_afrag(const __bf16* hrow, int k0, int koff) {
  v8bf lo = *(const v8bf*)(hrow + k0 + koff);        // 16B ds load
  v8bf hi = *(const v8bf*)(hrow + k0 + 16 + koff);   // 16B ds load
  return __builtin_shufflevector(lo, hi, 0, 1, 2, 3, 4, 5, 6, 7,
                                 8, 9, 10, 11, 12, 13, 14, 15);
}

__global__ __launch_bounds__(256, 1)
void lstm2_fused_kernel(const float* __restrict__ x,
                        const float* __restrict__ W_ih0, const float* __restrict__ W_hh0,
                        const float* __restrict__ b_ih0, const float* __restrict__ b_hh0,
                        const float* __restrict__ W_ih1, const float* __restrict__ W_hh1,
                        const float* __restrict__ b_ih1, const float* __restrict__ b_hh1,
                        const float* __restrict__ W_out, const float* __restrict__ b_out,
                        float* __restrict__ out) {
  __shared__ float  s_gact[BT][G4H + 8];   // activated gates (padded vs bank conflicts)
  __shared__ float  s_c[2][BT][H + 8];     // cell state, both layers, f32
  __shared__ float  s_hf[2][BT][H + 8];    // hidden state, f32 (head uses layer 1)
  __shared__ __align__(16) __bf16 s_hb[2][BT][H + 8]; // hidden state, bf16 (A frags)

  const int tid  = threadIdx.x;
  const int w    = tid >> 5;          // wave 0..7 -> gate columns [32w, 32w+32)
  const int lane = tid & 31;
  const int r    = lane & 15;         // matrix row this lane serves
  const int koff = (lane & 16) ? 8 : 0;
  const int mb   = (lane & 16) ? 8 : 0;   // C/D layout: lanes>=16 hold M=8..15
  const int gate = w >> 1;            // 0:i 1:f 2:g 3:o (wave-uniform)
  const int b0   = blockIdx.x * BT;

  // Branchless activation params (wave-uniform): act(x) = m*rcp(1+exp2(k*x)) + b
  //   sigmoid: k=-log2e,   m=1, b=0      tanh (gate g): k=-2*log2e, m=2, b=-1
  const float act_k = (gate == 2) ? (-2.0f * LOG2E) : (-LOG2E);
  const float act_m = (gate == 2) ?  2.0f :  1.0f;
  const float act_b = (gate == 2) ? -1.0f :  0.0f;

  // ---- zero-init persistent state -------------------------------------------
  for (int e = tid; e < 2 * BT * (H + 8); e += 256) {
    ((float*)s_c)[e]   = 0.0f;
    ((float*)s_hf)[e]  = 0.0f;
    ((__bf16*)s_hb)[e] = (__bf16)0.0f;
  }

  // ---- preload all weight B-fragments into VGPRs (loop-invariant) ------------
  v16bf Bih0[2];            // layer0 input proj, K=12 padded into one K-chunk
  v16bf Bhh0[2][2], Bih1[2][2], Bhh1[2][2];   // [n-tile][k-chunk]
  float bias0[2], bias1[2];
#pragma unroll
  for (int tI = 0; tI < 2; ++tI) {
    const int n0 = 32 * w + 16 * tI;
    Bih0[tI] = load_bfrag(W_ih0, DIN, n0, 0, lane);
#pragma unroll
    for (int kc = 0; kc < 2; ++kc) {
      Bhh0[tI][kc] = load_bfrag(W_hh0, H, n0, 32 * kc, lane);
      Bih1[tI][kc] = load_bfrag(W_ih1, H, n0, 32 * kc, lane);
      Bhh1[tI][kc] = load_bfrag(W_hh1, H, n0, 32 * kc, lane);
    }
    bias0[tI] = b_ih0[n0 + r] + b_hh0[n0 + r];
    bias1[tI] = b_ih1[n0 + r] + b_hh1[n0 + r];
  }
  __syncthreads();

  const float* xrow = x + (size_t)(b0 + r) * TLEN * DIN;

  for (int t = 0; t < TLEN; ++t) {
    if (w == 0 && lane < 16 && (t + 32) < TLEN)
      __builtin_prefetch(xrow + (size_t)(t + 32) * DIN, 0, 1);

    // =========================== layer 0 =====================================
    // A-fragment for x_t (16x12, zero-padded to 16x32); 48B rows are 16B aligned
    v16bf ax;
    {
      const float* xp = xrow + (size_t)t * DIN;
      if (lane < 16) {                       // k = 0..7, all valid
        float4 a = *(const float4*)(xp);
        float4 b = *(const float4*)(xp + 4);
        ax[0] = (__bf16)a.x; ax[1] = (__bf16)a.y; ax[2] = (__bf16)a.z; ax[3] = (__bf16)a.w;
        ax[4] = (__bf16)b.x; ax[5] = (__bf16)b.y; ax[6] = (__bf16)b.z; ax[7] = (__bf16)b.w;
      } else {                               // k = 8..11 valid, 12..15 zero
        float4 a = *(const float4*)(xp + 8);
        ax[0] = (__bf16)a.x; ax[1] = (__bf16)a.y; ax[2] = (__bf16)a.z; ax[3] = (__bf16)a.w;
        ax[4] = (__bf16)0.0f; ax[5] = (__bf16)0.0f; ax[6] = (__bf16)0.0f; ax[7] = (__bf16)0.0f;
      }
#pragma unroll
      for (int j = 8; j < 16; ++j) ax[j] = (__bf16)0.0f;   // K=16..31 pad
    }
    v16bf ah0 = load_afrag(&s_hb[0][r][0], 0, koff);
    v16bf ah1 = load_afrag(&s_hb[0][r][0], 32, koff);

#pragma unroll
    for (int tI = 0; tI < 2; ++tI) {
      v8f acc;
#pragma unroll
      for (int j = 0; j < 8; ++j) acc[j] = bias0[tI];      // bias uniform over M
      acc = wmma_bf16(ax,  Bih0[tI],    acc);
      acc = wmma_bf16(ah0, Bhh0[tI][0], acc);
      acc = wmma_bf16(ah1, Bhh0[tI][1], acc);
      const int col = 32 * w + 16 * tI + r;
#pragma unroll
      for (int g = 0; g < 8; ++g) {
        // branchless wave-uniform nonlinearity: m*rcp(1+exp2(k*x)) + b
        float v = fmaf(act_m, fast_rcp(1.0f + fast_exp2(act_k * acc[g])), act_b);
        s_gact[mb + g][col] = v;
      }
    }
    __syncthreads();   // gates ready; old h0 reads retired

    for (int e = tid; e < BT * H; e += 256) {              // cell update L0
      int rr = e >> 6, hd = e & 63;
      float iv = s_gact[rr][hd];
      float fv = s_gact[rr][64 + hd];
      float gv = s_gact[rr][128 + hd];
      float ov = s_gact[rr][192 + hd];
      float c = fv * s_c[0][rr][hd] + iv * gv;
      float h = ov * tanh_fast(c);
      s_c[0][rr][hd]  = c;
      s_hf[0][rr][hd] = h;
      s_hb[0][rr][hd] = (__bf16)h;
    }
    __syncthreads();   // new h0 visible

    // =========================== layer 1 =====================================
    v16bf ai0 = load_afrag(&s_hb[0][r][0], 0, koff);
    v16bf ai1 = load_afrag(&s_hb[0][r][0], 32, koff);
    v16bf ar0 = load_afrag(&s_hb[1][r][0], 0, koff);
    v16bf ar1 = load_afrag(&s_hb[1][r][0], 32, koff);

#pragma unroll
    for (int tI = 0; tI < 2; ++tI) {
      v8f acc;
#pragma unroll
      for (int j = 0; j < 8; ++j) acc[j] = bias1[tI];
      acc = wmma_bf16(ai0, Bih1[tI][0], acc);
      acc = wmma_bf16(ai1, Bih1[tI][1], acc);
      acc = wmma_bf16(ar0, Bhh1[tI][0], acc);
      acc = wmma_bf16(ar1, Bhh1[tI][1], acc);
      const int col = 32 * w + 16 * tI + r;
#pragma unroll
      for (int g = 0; g < 8; ++g) {
        float v = fmaf(act_m, fast_rcp(1.0f + fast_exp2(act_k * acc[g])), act_b);
        s_gact[mb + g][col] = v;
      }
    }
    __syncthreads();   // gates ready; old h1 reads retired

    for (int e = tid; e < BT * H; e += 256) {              // cell update L1
      int rr = e >> 6, hd = e & 63;
      float iv = s_gact[rr][hd];
      float fv = s_gact[rr][64 + hd];
      float gv = s_gact[rr][128 + hd];
      float ov = s_gact[rr][192 + hd];
      float c = fv * s_c[1][rr][hd] + iv * gv;
      float h = ov * tanh_fast(c);
      s_c[1][rr][hd]  = c;
      s_hf[1][rr][hd] = h;
      s_hb[1][rr][hd] = (__bf16)h;
    }
    __syncthreads();   // new h1 visible for next step
  }

  // ---- head: Linear(64,3) + LeakyReLU(0.01) on h1(T-1) -----------------------
  if (tid < BT * NO) {
    int rr = tid / NO, n = tid % NO;
    float s = b_out[n];
#pragma unroll
    for (int k = 0; k < H; ++k) s += s_hf[1][rr][k] * W_out[n * H + k];
    out[(size_t)(b0 + rr) * NO + n] = (s > 0.0f) ? s : 0.01f * s;
  }
}

extern "C" void kernel_launch(void* const* d_in, const int* in_sizes, int n_in,
                              void* d_out, int out_size, void* d_ws, size_t ws_size,
                              hipStream_t stream) {
  (void)in_sizes; (void)n_in; (void)d_ws; (void)ws_size; (void)out_size;
  const float* x     = (const float*)d_in[0];
  const float* W_ih0 = (const float*)d_in[1];
  const float* W_hh0 = (const float*)d_in[2];
  const float* b_ih0 = (const float*)d_in[3];
  const float* b_hh0 = (const float*)d_in[4];
  const float* W_ih1 = (const float*)d_in[5];
  const float* W_hh1 = (const float*)d_in[6];
  const float* b_ih1 = (const float*)d_in[7];
  const float* b_hh1 = (const float*)d_in[8];
  const float* W_out = (const float*)d_in[9];
  const float* b_out = (const float*)d_in[10];
  float* out = (float*)d_out;

  dim3 grid(BSZ / BT);   // 32 workgroups, one per 16-row batch tile
  dim3 block(256);       // 8 wave32s
  hipLaunchKernelGGL(lstm2_fused_kernel, grid, block, 0, stream,
                     x, W_ih0, W_hh0, b_ih0, b_hh0,
                     W_ih1, W_hh1, b_ih1, b_hh1, W_out, b_out, out);
}
